// SlidingWindowAttention_46669114638772
// MI455X (gfx1250) — matrix operand order repaired and checked by reading, not compiled, not run on hardware
//
#include <hip/hip_runtime.h>
#include <hip/hip_bf16.h>

typedef _Float16 f16;
typedef _Float16 v16h __attribute__((ext_vector_type(16)));
typedef _Float16 h8  __attribute__((ext_vector_type(8)));
typedef _Float16 h4  __attribute__((ext_vector_type(4)));
typedef _Float16 h2  __attribute__((ext_vector_type(2)));
typedef float    v8f __attribute__((ext_vector_type(8)));
typedef float    f4  __attribute__((ext_vector_type(4)));

#define B_   2
#define S_   4096
#define H_   16
#define HK_  4
#define D_   128
#define DIM_ 2048
#define MS_  (B_*S_)   /* 8192 rows */

static __device__ __forceinline__ v16h cat8(h8 lo, h8 hi) {
  return __builtin_shufflevector(lo, hi, 0,1,2,3,4,5,6,7,8,9,10,11,12,13,14,15);
}
static __device__ __forceinline__ v8f wmma16(v16h a, v16h b, v8f c) {
  // D(16x16 f32) = A(16x32 f16) * B(32x16 f16) + C
  return __builtin_amdgcn_wmma_f32_16x16x32_f16(false, a, false, b, (short)0, c, false, false);
}
// 20*tanh(s/(sqrt(128)*20)) via hardware exp+rcp: tanh(y)=sign(y)*(1-2/(e^{2|y|}+1))
static __device__ __forceinline__ float softcap20(float s) {
  const float sc = 0.0044194173824159216f;   // 1/(sqrt(128)*CAP)
  float y = s * sc;
  float e = __expf(fabsf(y) * 2.0f);
  float t = 1.0f - 2.0f * __builtin_amdgcn_rcpf(e + 1.0f);
  return copysignf(t, y) * 20.0f;
}

// ---------------------------------------------------------------- convert (8 elems/thread)
__global__ __launch_bounds__(256) void cvt_f32_f16(const float* __restrict__ in,
                                                   f16* __restrict__ out, int n8) {
  int i = blockIdx.x * 256 + threadIdx.x;
  if (i >= n8) return;
  const size_t b = (size_t)i * 8;
  f4 a0 = *(const f4*)(in + b);
  f4 a1 = *(const f4*)(in + b + 4);
  h8 r;
#pragma unroll
  for (int j = 0; j < 4; ++j) r[j] = (f16)a0[j];
#pragma unroll
  for (int j = 0; j < 4; ++j) r[4 + j] = (f16)a1[j];
  *(h8*)(out + b) = r;
}

// ---------------------------------------------------------------- RoPE (in place, f16)
// buf layout (B, S, nheads, 128); one thread per pair (elements 2i, 2i+1 via one b32)
__global__ __launch_bounds__(256) void rope_f16(f16* __restrict__ buf, int nheads, int npairs) {
  int idx = blockIdx.x * 256 + threadIdx.x;
  if (idx >= npairs) return;
  int i   = idx & 63;                       // pair index 0..63
  int pos = ((idx >> 6) / nheads) % S_;     // sequence position
  float inv = __expf(-(2.0f * (float)i / 128.0f) * 9.210340371976184f); // 10000^(-2i/128)
  float ang = (float)pos * inv;
  float c = __cosf(ang), s = __sinf(ang);   // v_cos/v_sin; error << f16 storage error
  h2* p = (h2*)(buf + (size_t)idx * 2);
  h2 xv = *p;
  float x1 = (float)xv[0], x2 = (float)xv[1];
  h2 r;
  r[0] = (f16)(x1 * c - x2 * s);
  r[1] = (f16)(x1 * s + x2 * c);
  *p = r;
}

// ---------------------------------------------------------------- GEMM  C = X(MxK) * W(NxK)^T
// Computed as C^T tiles (A = W rows, B = X^T) so stores are contiguous per lane.
// 256 threads = 8 waves (2 n-dim x 4 m-dim). Block tile 128x128, K-step 32.
#define BM 128
#define BN 128
#define BK 32

template <bool F32OUT>
__global__ __launch_bounds__(256) void gemm_nt(const f16* __restrict__ X,
                                               const f16* __restrict__ W,
                                               void* __restrict__ Cout,
                                               int M, int N, int K) {
  __shared__ f16 Xs[BM][BK];
  __shared__ f16 Ws[BN][BK];
  const int n0   = blockIdx.x * BN;
  const int m0   = blockIdx.y * BM;
  const int tid  = threadIdx.x;
  const int lane = tid & 31;
  const int w    = tid >> 5;
  const int wn   = w >> 2;          // 0..1  (64 n-rows each)
  const int wm   = w & 3;           // 0..3  (32 m-cols each)
  const int lm   = lane & 15;
  const int hi   = lane >> 4;

  const v8f zf = {0.f,0.f,0.f,0.f,0.f,0.f,0.f,0.f};
  v8f acc[4][2];                    // [n-tile][m-tile], C^T layout
#pragma unroll
  for (int i = 0; i < 4; ++i)
#pragma unroll
    for (int j = 0; j < 2; ++j) acc[i][j] = zf;

  const int srow = tid >> 1;          // 0..127
  const int skb  = (tid & 1) * 16;    // 0 or 16 halves

  for (int k0 = 0; k0 < K; k0 += BK) {
    *(v16h*)&Xs[srow][skb] = *(const v16h*)&X[(size_t)(m0 + srow) * K + k0 + skb];
    *(v16h*)&Ws[srow][skb] = *(const v16h*)&W[(size_t)(n0 + srow) * K + k0 + skb];
    if (k0 + BK < K) {
      __builtin_prefetch(&X[(size_t)(m0 + srow) * K + k0 + BK + skb], 0, 1);
      __builtin_prefetch(&W[(size_t)(n0 + srow) * K + k0 + BK + skb], 0, 1);
    }
    __syncthreads();

    v16h bx[2];                     // B = X^T fragments (lane col = m row)
#pragma unroll
    for (int t = 0; t < 2; ++t)
      bx[t] = *(const v16h*)&Xs[wm * 32 + t * 16 + lm][hi * 16];

#pragma unroll
    for (int in = 0; in < 4; ++in) {
      const int nr = wn * 64 + in * 16 + lm;   // A = W rows (lane = n row)
      h8 alo = *(const h8*)&Ws[nr][hi * 16];
      h8 ahi = *(const h8*)&Ws[nr][hi * 16 + 8];
      v16h aw = cat8(alo, ahi);
#pragma unroll
      for (int t = 0; t < 2; ++t) acc[in][t] = wmma16(aw, bx[t], acc[in][t]);
    }
    __syncthreads();
  }

  // store: C^T tile -> lane owns 8 consecutive n-columns of one m-row
#pragma unroll
  for (int in = 0; in < 4; ++in)
#pragma unroll
    for (int t = 0; t < 2; ++t) {
      const int gm  = m0 + wm * 32 + t * 16 + lm;            // m row (lane col)
      const int gn0 = n0 + wn * 64 + in * 16 + hi * 8;       // first of 8 consecutive n
      if (F32OUT) {
        float* C = (float*)Cout;
        f4 r0, r1;
#pragma unroll
        for (int v = 0; v < 4; ++v) { r0[v] = acc[in][t][v]; r1[v] = acc[in][t][4 + v]; }
        *(f4*)&C[(size_t)gm * N + gn0]     = r0;
        *(f4*)&C[(size_t)gm * N + gn0 + 4] = r1;
      } else {
        f16* C = (f16*)Cout;
        h8 r;
#pragma unroll
        for (int v = 0; v < 8; ++v) r[v] = (f16)acc[in][t][v];
        *(h8*)&C[(size_t)gm * N + gn0] = r;
      }
    }
}

// ---------------------------------------------------------------- attention (transposed)
// grid = (S/128, B*H), 256 threads. Wave w owns 16 queries (= lane columns).
// S^T = K*Q^T, flash softmax with per-lane scalar state, O^T = V^T * P^T.
__global__ __launch_bounds__(256) void attn_swa(const f16* __restrict__ qb,
                                                const f16* __restrict__ kb,
                                                const f16* __restrict__ vb,
                                                f16* __restrict__ ob) {
  __shared__ f16 Ks[32][128];        // [key][d]
  __shared__ f16 Vt[128][40];        // [d][key] (+pad)

  const int qt = blockIdx.x;
  const int bh = blockIdx.y;
  const int b  = bh / H_;
  const int h  = bh % H_;
  const int hk = h / (H_ / HK_);
  const int t0 = qt * 128;

  const int tid  = threadIdx.x;
  const int lane = tid & 31;
  const int w    = tid >> 5;
  const int lm   = lane & 15;
  const int hi   = lane >> 4;

  // ---- Q as B-fragments (lane column = query), 4 d-chunks, loaded once
  const int q = t0 + w * 16 + lm;
  const f16* qptr = qb + (((size_t)b * S_ + q) * H_ + h) * D_;
  v16h qf[4];
#pragma unroll
  for (int f = 0; f < 4; ++f)
    qf[f] = *(const v16h*)(qptr + f * 32 + hi * 16);

  const v8f zf = {0.f,0.f,0.f,0.f,0.f,0.f,0.f,0.f};
  v8f o[8];                           // O^T tiles: rows = d (t*16 + v + hi*8), col = query
#pragma unroll
  for (int t = 0; t < 8; ++t) o[t] = zf;
  float mrun = -1e30f, lrun = 0.f;    // per-query scalar state

  int klo = t0 - 512; if (klo < 0) klo = 0;
  const int khi = t0 + 127;

  const int skey = tid >> 3;             // K staging: 0..31
  const int sdb  = (tid & 7) * 16;       // 0..112 halves
  const int vkq  = (tid & 7) * 4;        // V staging: key group of 4
  const int vdq  = (tid >> 3) * 4;       // d group of 4 (0..124)

  for (int kc = klo; kc <= khi; kc += 32) {
    __syncthreads();
    {   // stage K row-major (16 halves = 32 B/thread)
      const f16* kp = kb + (((size_t)b * S_ + kc + skey) * HK_ + hk) * D_ + sdb;
      *(v16h*)&Ks[skey][sdb] = *(const v16h*)kp;
      // stage V transposed: 4 keys x 4 d per thread, packed b64 stores
      const f16* vp = vb + (((size_t)b * S_ + kc + vkq) * HK_ + hk) * D_ + vdq;
      h4 vr[4];
#pragma unroll
      for (int kk = 0; kk < 4; ++kk)
        vr[kk] = *(const h4*)(vp + (size_t)kk * (HK_ * D_));
#pragma unroll
      for (int j = 0; j < 4; ++j) {
        h4 cc;
#pragma unroll
        for (int kk = 0; kk < 4; ++kk) cc[kk] = vr[kk][j];
        *(h4*)&Vt[vdq + j][vkq] = cc;
      }
    }
    __syncthreads();

    // ---- S^T tiles: rows = keys (v + hi*8), col = query; reduce over d=128
    v8f s0 = zf, s1 = zf;
#pragma unroll
    for (int f = 0; f < 4; ++f) {
      h8 k0lo = *(const h8*)&Ks[lm][f * 32 + hi * 16];
      h8 k0hi = *(const h8*)&Ks[lm][f * 32 + hi * 16 + 8];
      h8 k1lo = *(const h8*)&Ks[16 + lm][f * 32 + hi * 16];
      h8 k1hi = *(const h8*)&Ks[16 + lm][f * 32 + hi * 16 + 8];
      s0 = wmma16(cat8(k0lo, k0hi), qf[f], s0);
      s1 = wmma16(cat8(k1lo, k1hi), qf[f], s1);
    }

    // ---- softcap + mask (per-lane: 16 keys for this query)
    float p0[8], p1[8];
    bool  ok0[8], ok1[8];
    float mx = -1e30f;
#pragma unroll
    for (int v = 0; v < 8; ++v) {
      const int key0 = kc + v + hi * 8;
      const int key1 = key0 + 16;
      float c0 = softcap20(s0[v]);
      float c1 = softcap20(s1[v]);
      ok0[v] = (key0 <= q) && (key0 >= q - 512);
      ok1[v] = (key1 <= q) && (key1 >= q - 512);
      p0[v] = ok0[v] ? c0 : -1e30f;
      p1[v] = ok1[v] ? c1 : -1e30f;
      mx = fmaxf(mx, fmaxf(p0[v], p1[v]));
    }
    mx = fmaxf(mx, __shfl_xor(mx, 16, 32));        // combine the two key-halves
    const float mnew = fmaxf(mrun, mx);
    const float fac  = __expf(mrun - mnew);
    float rs = 0.f;
#pragma unroll
    for (int v = 0; v < 8; ++v) {
      p0[v] = ok0[v] ? __expf(p0[v] - mnew) : 0.f;
      p1[v] = ok1[v] ? __expf(p1[v] - mnew) : 0.f;
      rs += p0[v] + p1[v];
    }
    rs += __shfl_xor(rs, 16, 32);
    lrun = lrun * fac + rs;
    mrun = mnew;
#pragma unroll
    for (int t = 0; t < 8; ++t) o[t] = o[t] * fac;

    // ---- build P^T B-fragment via one half-wave exchange (no LDS round-trip)
    float recv[8];
#pragma unroll
    for (int v = 0; v < 8; ++v) {
      float send = hi ? p0[v] : p1[v];
      recv[v] = __shfl_xor(send, 16, 32);
    }
    v16h pf;
#pragma unroll
    for (int e = 0; e < 8; ++e) pf[e]     = (f16)(hi ? recv[e] : p0[e]);
#pragma unroll
    for (int e = 0; e < 8; ++e) pf[8 + e] = (f16)(hi ? p1[e] : recv[e]);

    // ---- O^T += V^T(128 x 32keys) * P^T(32keys x 16q)
#pragma unroll
    for (int t = 0; t < 8; ++t) {
      h8 vlo = *(const h8*)&Vt[t * 16 + lm][hi * 16];
      h8 vhi = *(const h8*)&Vt[t * 16 + lm][hi * 16 + 8];
      o[t] = wmma16(cat8(vlo, vhi), pf, o[t]);
    }
  }

  // ---- epilogue: per lane 8 consecutive d per tile -> vector stores
  const float invl = __builtin_amdgcn_rcpf(lrun);
  f16* optr = ob + (((size_t)b * S_ + q) * H_ + h) * D_;
#pragma unroll
  for (int t = 0; t < 8; ++t) {
    h8 r;
#pragma unroll
    for (int v = 0; v < 8; ++v) r[v] = (f16)(o[t][v] * invl);
    *(h8*)(optr + t * 16 + hi * 8) = r;
  }
}

// ---------------------------------------------------------------- launch
extern "C" void kernel_launch(void* const* d_in, const int* in_sizes, int n_in,
                              void* d_out, int out_size, void* d_ws, size_t ws_size,
                              hipStream_t stream) {
  const float* x  = (const float*)d_in[0];
  const float* wq = (const float*)d_in[1];
  const float* wk = (const float*)d_in[2];
  const float* wv = (const float*)d_in[3];
  const float* wo = (const float*)d_in[4];

  char* ws = (char*)d_ws;
  size_t off = 0;
  auto alloc = [&](size_t bytes) -> void* {
    void* p = ws + off;
    off += (bytes + 255) & ~(size_t)255;
    return p;
  };
  f16* xh  = (f16*)alloc((size_t)MS_ * DIM_ * 2);   // reused as attention output
  f16* wqh = (f16*)alloc((size_t)DIM_ * DIM_ * 2);
  f16* wkh = (f16*)alloc((size_t)(HK_*D_) * DIM_ * 2);
  f16* wvh = (f16*)alloc((size_t)(HK_*D_) * DIM_ * 2);
  f16* woh = (f16*)alloc((size_t)DIM_ * DIM_ * 2);
  f16* qbuf = (f16*)alloc((size_t)MS_ * DIM_ * 2);
  f16* kbuf = (f16*)alloc((size_t)MS_ * (HK_*D_) * 2);
  f16* vbuf = (f16*)alloc((size_t)MS_ * (HK_*D_) * 2);
  f16* abuf = xh;  // x is dead after QKV projections

  auto cvt = [&](const float* src, f16* dst, int n) {
    int n8 = n / 8;
    cvt_f32_f16<<<(n8 + 255) / 256, 256, 0, stream>>>(src, dst, n8);
  };
  cvt(x,  xh,  MS_ * DIM_);
  cvt(wq, wqh, DIM_ * DIM_);
  cvt(wk, wkh, HK_ * D_ * DIM_);
  cvt(wv, wvh, HK_ * D_ * DIM_);
  cvt(wo, woh, DIM_ * DIM_);

  // Q/K/V projections (C = X * W^T), f16 outputs
  gemm_nt<false><<<dim3(DIM_ / BN, MS_ / BM), 256, 0, stream>>>(xh, wqh, qbuf, MS_, DIM_, DIM_);
  gemm_nt<false><<<dim3((HK_*D_) / BN, MS_ / BM), 256, 0, stream>>>(xh, wkh, kbuf, MS_, HK_*D_, DIM_);
  gemm_nt<false><<<dim3((HK_*D_) / BN, MS_ / BM), 256, 0, stream>>>(xh, wvh, vbuf, MS_, HK_*D_, DIM_);

  // RoPE in place
  {
    int npq = MS_ * H_ * (D_ / 2);
    int npk = MS_ * HK_ * (D_ / 2);
    rope_f16<<<(npq + 255) / 256, 256, 0, stream>>>(qbuf, H_,  npq);
    rope_f16<<<(npk + 255) / 256, 256, 0, stream>>>(kbuf, HK_, npk);
  }

  // sliding-window attention
  attn_swa<<<dim3(S_ / 128, B_ * H_), 256, 0, stream>>>(qbuf, kbuf, vbuf, abuf);

  // output projection, f32 into d_out
  gemm_nt<true><<<dim3(DIM_ / BN, MS_ / BM), 256, 0, stream>>>(abuf, woh, d_out, MS_, DIM_, DIM_);
}
